// mol_residues_light_level_encoder_10058813407606
// MI455X (gfx1250) — matrix operand order repaired and checked by reading, not compiled, not run on hardware
//
#include <hip/hip_runtime.h>
#include <hip/hip_bf16.h>

#define NTOK   4096
#define BATCH  4
#define NMAXI  1024
#define DIM    128
#define NH     4
#define DHD    32
#define NL     4
#define ECNT   8192
#define ETOTC  73728
#define DRES   112

typedef __attribute__((ext_vector_type(16))) __bf16 v16bf;
typedef __attribute__((ext_vector_type(8)))  float  v8f;

__device__ __forceinline__ __bf16 f2bf(float f) {
  union { float f; unsigned u; } v; v.f = f;
  unsigned r = v.u + 0x7FFFu + ((v.u >> 16) & 1u);   // round-to-nearest-even
  unsigned short hs = (unsigned short)(r >> 16);
  __bf16 o; __builtin_memcpy(&o, &hs, 2);
  return o;
}

// ---------------------------------------------------------------------------
// Generic bf16 WMMA GEMM: C[M,N] = alpha * A[M,K] @ Bt[N,K]^T + bias, optional
// ReLU, output f32 or bf16.
// Register-blocked: one wave computes a 32x32 macro-tile (2x2 grid of 16x16
// WMMA tiles) -> 4 independent v_wmma_f32_16x16x32_bf16 per K-step, 2x reuse
// of each loaded fragment, 4 accumulator chains to hide WMMA latency.
// Fragment layouts per CDNA5 ISA 7.12.2 (wave32).
// ---------------------------------------------------------------------------
template <bool RELU, bool OUTBF>
__global__ void gemm_wmma_bf16(const __bf16* __restrict__ A, int lda,
                               const __bf16* __restrict__ Bt, int ldb,
                               const float* __restrict__ bias, float alpha,
                               float* __restrict__ C, __bf16* __restrict__ Cb,
                               int ldc, int M, int Nout, int K) {
  const int tilesN = Nout >> 5;                 // 32-wide macro tiles
  const int tiles  = (M >> 5) * tilesN;
  const int tile   = blockIdx.x * 8 + (threadIdx.x >> 5);
  if (tile >= tiles) return;
  const int tM   = tile / tilesN;
  const int tN   = tile - tM * tilesN;
  const int lane = threadIdx.x & 31;
  const int half = lane >> 4;     // half-wave select
  const int mr   = lane & 15;     // A: row within tile, B: col within tile

  const __bf16* Ar0 = A  + (size_t)(tM * 32 + mr) * lda;
  const __bf16* Ar1 = Ar0 + (size_t)16 * lda;
  const __bf16* Br0 = Bt + (size_t)(tN * 32 + mr) * ldb;
  const __bf16* Br1 = Br0 + (size_t)16 * ldb;

  v8f acc00 = {}, acc01 = {}, acc10 = {}, acc11 = {};
  for (int kk = 0; kk < K; kk += 32) {
    if (kk + 32 < K) {                          // prefetch next K slice (L2 pipelining)
      __builtin_prefetch(Ar0 + kk + 32, 0, 3);
      __builtin_prefetch(Ar1 + kk + 32, 0, 3);
      __builtin_prefetch(Br0 + kk + 32, 0, 3);
      __builtin_prefetch(Br1 + kk + 32, 0, 3);
    }
    v16bf a0, a1, b0, b1;
#pragma unroll
    for (int v = 0; v < 8; ++v) {
      // A 16x32 bf16 layout: lanes 0-15 M, VGPR v -> K = (v&4?16:0)+half*8+(v&3)*2
      const int ka = kk + ((v & 4) ? 16 : 0) + half * 8 + (v & 3) * 2;
      a0[2 * v] = Ar0[ka]; a0[2 * v + 1] = Ar0[ka + 1];
      a1[2 * v] = Ar1[ka]; a1[2 * v + 1] = Ar1[ka + 1];
      // B 32x16 bf16 layout: lanes 0-15 N, VGPR v -> K = (v&4?16:0)+half*8+(v&3)*2 (same as A)
      const int kb = kk + ((v & 4) ? 16 : 0) + half * 8 + (v & 3) * 2;
      b0[2 * v] = Br0[kb]; b0[2 * v + 1] = Br0[kb + 1];
      b1[2 * v] = Br1[kb]; b1[2 * v + 1] = Br1[kb + 1];
    }
    acc00 = __builtin_amdgcn_wmma_f32_16x16x32_bf16(false, a0, false, b0, (short)0, acc00, false, false);
    acc01 = __builtin_amdgcn_wmma_f32_16x16x32_bf16(false, a0, false, b1, (short)0, acc01, false, false);
    acc10 = __builtin_amdgcn_wmma_f32_16x16x32_bf16(false, a1, false, b0, (short)0, acc10, false, false);
    acc11 = __builtin_amdgcn_wmma_f32_16x16x32_bf16(false, a1, false, b1, (short)0, acc11, false, false);
  }

  const int col0 = tN * 32 + mr;
  const int col1 = col0 + 16;
  const float bv0 = bias ? bias[col0] : 0.0f;
  const float bv1 = bias ? bias[col1] : 0.0f;
#pragma unroll
  for (int r = 0; r < 8; ++r) {
    const int row0 = tM * 32 + half * 8 + r;    // C layout: VGPR r -> M = r (+8 hi half)
    const int row1 = row0 + 16;
    float v00 = acc00[r] * alpha + bv0;
    float v01 = acc01[r] * alpha + bv1;
    float v10 = acc10[r] * alpha + bv0;
    float v11 = acc11[r] * alpha + bv1;
    if (RELU) {
      v00 = fmaxf(v00, 0.0f); v01 = fmaxf(v01, 0.0f);
      v10 = fmaxf(v10, 0.0f); v11 = fmaxf(v11, 0.0f);
    }
    if (OUTBF) {
      Cb[(size_t)row0 * ldc + col0] = f2bf(v00);
      Cb[(size_t)row0 * ldc + col1] = f2bf(v01);
      Cb[(size_t)row1 * ldc + col0] = f2bf(v10);
      Cb[(size_t)row1 * ldc + col1] = f2bf(v11);
    } else {
      C[(size_t)row0 * ldc + col0] = v00;
      C[(size_t)row0 * ldc + col1] = v01;
      C[(size_t)row1 * ldc + col0] = v10;
      C[(size_t)row1 * ldc + col1] = v11;
    }
  }
}

// ---------------------------------------------------------------------------
// Weight convert + transpose: W[l][In][Out] f32 -> Wt[l][Out][In] bf16
// ---------------------------------------------------------------------------
__global__ void convt_kernel(const float* __restrict__ W, __bf16* __restrict__ Wt,
                             int In, int Out) {
  const int l = blockIdx.y;
  const int t = blockIdx.x * 256 + threadIdx.x;
  if (t >= In * Out) return;
  const int i = t / Out, o = t - i * Out;
  Wt[(size_t)l * In * Out + (size_t)o * In + i] = f2bf(W[(size_t)l * In * Out + t]);
}

// ---------------------------------------------------------------------------
// Node embedding: residue tables + SASA linear (last 16 dims) + positional +
// chain embeddings. Writes f32 h and bf16 hb.
// ---------------------------------------------------------------------------
__global__ void embed_kernel(const int* __restrict__ x, const float* __restrict__ sasa,
                             const int* __restrict__ rid, const int* __restrict__ cid,
                             const float* __restrict__ rt0, const float* __restrict__ rt1,
                             const float* __restrict__ rt2, const float* __restrict__ sw,
                             const float* __restrict__ sb, const float* __restrict__ pos,
                             const float* __restrict__ chain,
                             float* __restrict__ h, __bf16* __restrict__ hb) {
  const int t = blockIdx.x * 256 + threadIdx.x;       // (node, dword4)
  if (t >= NTOK * 32) return;
  const int n = t >> 5, d4 = t & 31, d = d4 * 4;
  float v[4];
  if (d4 < 28) {                                      // residue embedding part (112 dims)
    const int x0 = x[3 * n], x1 = x[3 * n + 1], x2 = x[3 * n + 2];
#pragma unroll
    for (int j = 0; j < 4; ++j)
      v[j] = rt0[(size_t)x0 * DRES + d + j] + rt1[(size_t)x1 * DRES + d + j]
           + rt2[(size_t)x2 * DRES + d + j];
  } else {                                            // SASA linear part (16 dims)
    const int dd = d - DRES;
    float s0 = sasa[4 * n], s1 = sasa[4 * n + 1], s2 = sasa[4 * n + 2], s3 = sasa[4 * n + 3];
    s0 = (s0 != s0) ? 0.0f : s0;  s1 = (s1 != s1) ? 0.0f : s1;
    s2 = (s2 != s2) ? 0.0f : s2;  s3 = (s3 != s3) ? 0.0f : s3;
#pragma unroll
    for (int j = 0; j < 4; ++j)
      v[j] = sb[dd + j] + s0 * sw[dd + j] + s1 * sw[16 + dd + j]
           + s2 * sw[32 + dd + j] + s3 * sw[48 + dd + j];
  }
  const int r = rid[n], c = cid[n];
#pragma unroll
  for (int j = 0; j < 4; ++j) {
    const float vv = v[j] + pos[(size_t)r * DIM + d + j] + chain[(size_t)c * DIM + d + j];
    h [(size_t)n * DIM + d + j] = vv;
    hb[(size_t)n * DIM + d + j] = f2bf(vv);
  }
}

// ---------------------------------------------------------------------------
// GINE aggregation: agg[dst] += relu(h[src] + edge_embed(e)). One wave/edge,
// 4 dims per lane, edge embedding gathered on the fly (no ea buffer).
// ---------------------------------------------------------------------------
__global__ void gine_agg_kernel(const int* __restrict__ ei, const float* __restrict__ h,
                                const int* __restrict__ ca, const int* __restrict__ na,
                                const float* __restrict__ c0, const float* __restrict__ c1,
                                const float* __restrict__ n0, const float* __restrict__ n1,
                                float* __restrict__ agg) {
  const int e = blockIdx.x * 8 + (threadIdx.x >> 5);
  if (e >= ETOTC) return;
  const int lane = threadIdx.x & 31;
  const int src = ei[e];
  const int dst = ei[ETOTC + e];
  const float *t0, *t1;
  if (e < ECNT) {
    t0 = c0 + (size_t)ca[2 * e] * DIM;
    t1 = c1 + (size_t)ca[2 * e + 1] * DIM;
  } else {
    const int e2 = e - ECNT;
    t0 = n0 + (size_t)na[2 * e2] * DIM;
    t1 = n1 + (size_t)na[2 * e2 + 1] * DIM;
  }
  const float* hs = h + (size_t)src * DIM;
  float* ad = agg + (size_t)dst * DIM;
#pragma unroll
  for (int j = 0; j < 4; ++j) {
    const int d = lane * 4 + j;
    const float v = fmaxf(hs[d] + t0[d] + t1[d], 0.0f);
    atomicAdd(ad + d, v);
  }
}

__global__ void gine_pre_kernel(const float* __restrict__ h, const float* __restrict__ agg,
                                const float* __restrict__ epsp, __bf16* __restrict__ gA) {
  const int t = blockIdx.x * 256 + threadIdx.x;
  if (t >= NTOK * DIM) return;
  gA[t] = f2bf((1.0f + epsp[0]) * h[t] + agg[t]);
}

// ---------------------------------------------------------------------------
// Row LayerNorm over D=128 of (a+b), affine; optional fused bf16 output.
// ---------------------------------------------------------------------------
__global__ void ln_residual_kernel(const float* __restrict__ a, const float* __restrict__ b,
                                   const float* __restrict__ g, const float* __restrict__ be,
                                   float* __restrict__ out, __bf16* __restrict__ outb) {
  const int row = blockIdx.x;
  const int d = threadIdx.x;                     // 128 threads
  const float z = a[(size_t)row * DIM + d] + b[(size_t)row * DIM + d];
  __shared__ float rs[DIM], rq[DIM];
  rs[d] = z; rq[d] = z * z;
  __syncthreads();
  for (int s = 64; s > 0; s >>= 1) {
    if (d < s) { rs[d] += rs[d + s]; rq[d] += rq[d + s]; }
    __syncthreads();
  }
  const float mu  = rs[0] * (1.0f / DIM);
  const float var = rq[0] * (1.0f / DIM) - mu * mu;
  const float o = (z - mu) * rsqrtf(var + 1e-5f) * g[d] + be[d];
  out[(size_t)row * DIM + d] = o;
  if (outb) outb[(size_t)row * DIM + d] = f2bf(o);
}

// ---------------------------------------------------------------------------
// Row softmax over 1024, fused hop-decay bias 0.5^max(d3d-hop,0) and bf16 out.
// ---------------------------------------------------------------------------
__global__ void softmax_sph_kernel(const float* __restrict__ S, __bf16* __restrict__ P,
                                   const float* __restrict__ d3d,
                                   const float* __restrict__ hop, int hh) {
  const int row = blockIdx.x, t = threadIdx.x;   // 256 threads, 4 elems each
  const float* Sr = S + (size_t)row * NMAXI;
  float v0 = Sr[t * 4 + 0], v1 = Sr[t * 4 + 1], v2 = Sr[t * 4 + 2], v3 = Sr[t * 4 + 3];
  __shared__ float red[256];
  red[t] = fmaxf(fmaxf(v0, v1), fmaxf(v2, v3));
  __syncthreads();
  for (int s = 128; s > 0; s >>= 1) {
    if (t < s) red[t] = fmaxf(red[t], red[t + s]);
    __syncthreads();
  }
  const float m = red[0];
  __syncthreads();
  const float e0 = __expf(v0 - m), e1 = __expf(v1 - m), e2 = __expf(v2 - m), e3 = __expf(v3 - m);
  red[t] = e0 + e1 + e2 + e3;
  __syncthreads();
  for (int s = 128; s > 0; s >>= 1) {
    if (t < s) red[t] += red[t + s];
    __syncthreads();
  }
  const float inv = 1.0f / red[0];
  const float hv  = hop[hh];
  const float* dr = d3d + (size_t)row * NMAXI;
  const float e[4] = {e0, e1, e2, e3};
#pragma unroll
  for (int j = 0; j < 4; ++j) {
    const int c = t * 4 + j;
    const float sph = exp2f(-fmaxf(dr[c] - hv, 0.0f));   // 0.5^LeakyReLU(d-hop), slope 0
    P[(size_t)row * NMAXI + c] = f2bf(e[j] * inv * sph);
  }
}

// V row-major bf16 [4096,128] -> Vt[(b*NH+h)*DH + d][NMAXI] bf16 (K-contig for WMMA B)
__global__ void vtrans_kernel(const __bf16* __restrict__ Vr, __bf16* __restrict__ Vt) {
  const int t = blockIdx.x * 256 + threadIdx.x;
  if (t >= NTOK * DIM) return;
  const int mg = t >> 7, c = t & 127;
  const int b = mg >> 10, m = mg & 1023;
  const int hh = c >> 5, d = c & 31;
  Vt[((size_t)(b * NH + hh) * DHD + d) * NMAXI + m] = Vr[t];
}

__global__ void combine_kernel(const float* __restrict__ a, const float* __restrict__ b,
                               float* __restrict__ o, __bf16* __restrict__ ob) {
  const int t = blockIdx.x * 256 + threadIdx.x;
  if (t >= NTOK * DIM) return;
  const float v = a[t] + b[t];
  o[t] = v; ob[t] = f2bf(v);
}

__global__ void cvt_bf16_kernel(const float* __restrict__ a, __bf16* __restrict__ o, int n) {
  const int t = blockIdx.x * 256 + threadIdx.x;
  if (t < n) o[t] = f2bf(a[t]);
}

__global__ void fill_ones_kernel(float* __restrict__ p, int n) {
  const int t = blockIdx.x * 256 + threadIdx.x;
  if (t < n) p[t] = 1.0f;
}

// ---------------------------------------------------------------------------
extern "C" void kernel_launch(void* const* d_in, const int* in_sizes, int n_in,
                              void* d_out, int out_size, void* d_ws, size_t ws_size,
                              hipStream_t stream) {
  (void)in_sizes; (void)n_in; (void)ws_size;
  const int*   x       = (const int*)  d_in[0];
  const float* sasa    = (const float*)d_in[1];
  const int*   rid     = (const int*)  d_in[3];
  const int*   cid     = (const int*)  d_in[4];
  const int*   ei      = (const int*)  d_in[5];
  const int*   covattr = (const int*)  d_in[7];
  const int*   ncattr  = (const int*)  d_in[10];
  const float* d3d     = (const float*)d_in[12];
  const float* rt0     = (const float*)d_in[14];
  const float* rt1     = (const float*)d_in[15];
  const float* rt2     = (const float*)d_in[16];
  const float* sw      = (const float*)d_in[17];
  const float* sbv     = (const float*)d_in[18];
  const float* pos     = (const float*)d_in[19];
  const float* chain   = (const float*)d_in[20];
  const float* cov0    = (const float*)d_in[21];
  const float* cov1    = (const float*)d_in[22];
  const float* nc0     = (const float*)d_in[23];
  const float* nc1     = (const float*)d_in[24];
  const float* hop     = (const float*)d_in[25];
  const float* geps    = (const float*)d_in[26];
  const float* gw1     = (const float*)d_in[27];
  const float* gb1     = (const float*)d_in[28];
  const float* gw2     = (const float*)d_in[29];
  const float* gb2     = (const float*)d_in[30];
  const float* wq      = (const float*)d_in[31];
  const float* wk      = (const float*)d_in[32];
  const float* wv      = (const float*)d_in[33];
  const float* wo      = (const float*)d_in[34];
  const float* bq      = (const float*)d_in[35];
  const float* bk      = (const float*)d_in[36];
  const float* bvv     = (const float*)d_in[37];
  const float* bo      = (const float*)d_in[38];
  const float* g1      = (const float*)d_in[39];
  const float* g2      = (const float*)d_in[40];
  const float* g3      = (const float*)d_in[41];
  const float* be1     = (const float*)d_in[42];
  const float* be2     = (const float*)d_in[43];
  const float* be3     = (const float*)d_in[44];
  const float* fw1     = (const float*)d_in[45];
  const float* fb1     = (const float*)d_in[46];
  const float* fw2     = (const float*)d_in[47];
  const float* fb2     = (const float*)d_in[48];

  // ---- workspace carve-out (~35 MB) ----
  char* base = (char*)d_ws; size_t off = 0;
  auto alloc = [&](size_t bytes) -> void* {
    void* p = base + off;
    off = (off + bytes + 255) & ~(size_t)255;
    return p;
  };
  float*  h    = (float*) alloc((size_t)NTOK * DIM * 4);
  __bf16* hb   = (__bf16*)alloc((size_t)NTOK * DIM * 2);
  float*  agg  = (float*) alloc((size_t)NTOK * DIM * 4);
  __bf16* gA   = (__bf16*)alloc((size_t)NTOK * DIM * 2);
  __bf16* t1b  = (__bf16*)alloc((size_t)NTOK * 2 * DIM * 2);
  float*  t2   = (float*) alloc((size_t)NTOK * DIM * 4);
  float*  locb = (float*) alloc((size_t)NTOK * DIM * 4);
  float*  attb = (float*) alloc((size_t)NTOK * DIM * 4);
  float*  outc = (float*) alloc((size_t)NTOK * DIM * 4);
  __bf16* outb = (__bf16*)alloc((size_t)NTOK * DIM * 2);
  __bf16* Qb   = (__bf16*)alloc((size_t)NTOK * DIM * 2);
  __bf16* Kb   = (__bf16*)alloc((size_t)NTOK * DIM * 2);
  __bf16* Vr   = (__bf16*)alloc((size_t)NTOK * DIM * 2);
  __bf16* Vt   = (__bf16*)alloc((size_t)NTOK * DIM * 2);
  float*  Sbuf = (float*) alloc((size_t)NMAXI * NMAXI * 4);
  __bf16* Pb   = (__bf16*)alloc((size_t)NMAXI * NMAXI * 2);
  float*  obuf = (float*) alloc((size_t)NTOK * DIM * 4);
  __bf16* ob   = (__bf16*)alloc((size_t)NTOK * DIM * 2);
  __bf16* gw1t = (__bf16*)alloc((size_t)NL * DIM * DIM * 2);
  __bf16* gw2t = (__bf16*)alloc((size_t)NL * DIM * DIM * 2);
  __bf16* wqt  = (__bf16*)alloc((size_t)NL * DIM * DIM * 2);
  __bf16* wkt  = (__bf16*)alloc((size_t)NL * DIM * DIM * 2);
  __bf16* wvt  = (__bf16*)alloc((size_t)NL * DIM * DIM * 2);
  __bf16* wot  = (__bf16*)alloc((size_t)NL * DIM * DIM * 2);
  __bf16* fw1t = (__bf16*)alloc((size_t)NL * DIM * 2 * DIM * 2);
  __bf16* fw2t = (__bf16*)alloc((size_t)NL * 2 * DIM * DIM * 2);

  auto launch_gemm = [&](const __bf16* A, int lda, const __bf16* Bt, int ldb,
                         const float* bias, float alpha, float* C, __bf16* Cb,
                         int ldc, int M, int Nn, int K, bool relu, bool outbf) {
    const int tiles  = (M >> 5) * (Nn >> 5);    // 32x32 macro tiles, one per wave
    const int blocks = (tiles + 7) / 8;
    if (relu) {
      if (outbf) gemm_wmma_bf16<true, true ><<<blocks, 256, 0, stream>>>(A, lda, Bt, ldb, bias, alpha, C, Cb, ldc, M, Nn, K);
      else       gemm_wmma_bf16<true, false><<<blocks, 256, 0, stream>>>(A, lda, Bt, ldb, bias, alpha, C, Cb, ldc, M, Nn, K);
    } else {
      if (outbf) gemm_wmma_bf16<false, true ><<<blocks, 256, 0, stream>>>(A, lda, Bt, ldb, bias, alpha, C, Cb, ldc, M, Nn, K);
      else       gemm_wmma_bf16<false, false><<<blocks, 256, 0, stream>>>(A, lda, Bt, ldb, bias, alpha, C, Cb, ldc, M, Nn, K);
    }
  };

  // ---- one-time weight convert+transpose (bf16, [out][in]) ----
  {
    dim3 gsq((DIM * DIM + 255) / 256, NL);
    convt_kernel<<<gsq, 256, 0, stream>>>(gw1, gw1t, DIM, DIM);
    convt_kernel<<<gsq, 256, 0, stream>>>(gw2, gw2t, DIM, DIM);
    convt_kernel<<<gsq, 256, 0, stream>>>(wq,  wqt,  DIM, DIM);
    convt_kernel<<<gsq, 256, 0, stream>>>(wk,  wkt,  DIM, DIM);
    convt_kernel<<<gsq, 256, 0, stream>>>(wv,  wvt,  DIM, DIM);
    convt_kernel<<<gsq, 256, 0, stream>>>(wo,  wot,  DIM, DIM);
    dim3 gr1((DIM * 2 * DIM + 255) / 256, NL);
    convt_kernel<<<gr1, 256, 0, stream>>>(fw1, fw1t, DIM, 2 * DIM);
    convt_kernel<<<gr1, 256, 0, stream>>>(fw2, fw2t, 2 * DIM, DIM);
  }

  // ---- node embeddings ----
  embed_kernel<<<(NTOK * 32 + 255) / 256, 256, 0, stream>>>(
      x, sasa, rid, cid, rt0, rt1, rt2, sw, sbv, pos, chain, h, hb);

  const float scale = 0.17677669529663687f;  // 1/sqrt(32)

  for (int l = 0; l < NL; ++l) {
    // --- GINE local branch ---
    hipMemsetAsync(agg, 0, (size_t)NTOK * DIM * 4, stream);
    gine_agg_kernel<<<ETOTC / 8, 256, 0, stream>>>(ei, h, covattr, ncattr,
                                                   cov0, cov1, nc0, nc1, agg);
    gine_pre_kernel<<<(NTOK * DIM + 255) / 256, 256, 0, stream>>>(h, agg, geps + l, gA);
    launch_gemm(gA,  DIM, gw1t + (size_t)l * DIM * DIM, DIM, gb1 + l * DIM, 1.0f,
                nullptr, t1b, DIM, NTOK, DIM, DIM, true,  true);
    launch_gemm(t1b, DIM, gw2t + (size_t)l * DIM * DIM, DIM, gb2 + l * DIM, 1.0f,
                t2, nullptr, DIM, NTOK, DIM, DIM, false, false);
    ln_residual_kernel<<<NTOK, DIM, 0, stream>>>(t2, h, g1 + l * DIM, be1 + l * DIM,
                                                 locb, nullptr);

    // --- attention branch ---
    launch_gemm(hb, DIM, wqt + (size_t)l * DIM * DIM, DIM, bq + l * DIM, 1.0f,
                nullptr, Qb, DIM, NTOK, DIM, DIM, false, true);
    launch_gemm(hb, DIM, wkt + (size_t)l * DIM * DIM, DIM, bk + l * DIM, 1.0f,
                nullptr, Kb, DIM, NTOK, DIM, DIM, false, true);
    launch_gemm(hb, DIM, wvt + (size_t)l * DIM * DIM, DIM, bvv + l * DIM, 1.0f,
                nullptr, Vr, DIM, NTOK, DIM, DIM, false, true);
    vtrans_kernel<<<(NTOK * DIM + 255) / 256, 256, 0, stream>>>(Vr, Vt);

    for (int b = 0; b < BATCH; ++b) {
      for (int hh = 0; hh < NH; ++hh) {
        const __bf16* Qs = Qb + (size_t)b * NMAXI * DIM + hh * DHD;
        const __bf16* Ks = Kb + (size_t)b * NMAXI * DIM + hh * DHD;
        // S = (Q @ K^T) * scale : Bt == row-major K
        launch_gemm(Qs, DIM, Ks, DIM, nullptr, scale, Sbuf, nullptr,
                    NMAXI, NMAXI, NMAXI, DHD, false, false);
        softmax_sph_kernel<<<NMAXI, 256, 0, stream>>>(
            Sbuf, Pb, d3d + (size_t)b * NMAXI * NMAXI, hop, hh);
        const __bf16* Vs = Vt + (size_t)(b * NH + hh) * DHD * NMAXI;
        launch_gemm(Pb, NMAXI, Vs, NMAXI, nullptr, 1.0f,
                    obuf + (size_t)b * NMAXI * DIM + hh * DHD, nullptr,
                    DIM, NMAXI, DHD, NMAXI, false, false);
      }
    }
    cvt_bf16_kernel<<<(NTOK * DIM + 255) / 256, 256, 0, stream>>>(obuf, ob, NTOK * DIM);
    launch_gemm(ob, DIM, wot + (size_t)l * DIM * DIM, DIM, bo + l * DIM, 1.0f,
                t2, nullptr, DIM, NTOK, DIM, DIM, false, false);
    ln_residual_kernel<<<NTOK, DIM, 0, stream>>>(t2, h, g2 + l * DIM, be2 + l * DIM,
                                                 attb, nullptr);

    // --- combine + FFN ---
    combine_kernel<<<(NTOK * DIM + 255) / 256, 256, 0, stream>>>(locb, attb, outc, outb);
    launch_gemm(outb, DIM, fw1t + (size_t)l * DIM * 2 * DIM, DIM, fb1 + l * 2 * DIM, 1.0f,
                nullptr, t1b, 2 * DIM, NTOK, 2 * DIM, DIM, true, true);
    launch_gemm(t1b, 2 * DIM, fw2t + (size_t)l * 2 * DIM * DIM, 2 * DIM, fb2 + l * DIM, 1.0f,
                t2, nullptr, DIM, NTOK, DIM, 2 * DIM, false, false);
    ln_residual_kernel<<<NTOK, DIM, 0, stream>>>(outc, t2, g3 + l * DIM, be3 + l * DIM,
                                                 h, hb);
  }

  // ---- outputs: h [B,NMAX,D] then all-ones mask ----
  hipMemcpyAsync(d_out, h, (size_t)NTOK * DIM * 4, hipMemcpyDeviceToDevice, stream);
  const int rem = out_size - NTOK * DIM;
  if (rem > 0)
    fill_ones_kernel<<<(rem + 255) / 256, 256, 0, stream>>>((float*)d_out + NTOK * DIM, rem);
}
